// GranMultiHeadAttention_50070728737037
// MI455X (gfx1250) — matrix operand order repaired and checked by reading, not compiled, not run on hardware
//
#include <hip/hip_runtime.h>
#include <hip/hip_bf16.h>

// GranMultiHeadAttention for MI455X (gfx1250, wave32, WMMA f32_16x16x32_f16)

typedef __attribute__((ext_vector_type(16))) _Float16 v16h;
typedef __attribute__((ext_vector_type(8)))  _Float16 v8h;
typedef __attribute__((ext_vector_type(8)))  float    v8f;
typedef __attribute__((ext_vector_type(4)))  float    v4f;

#define KS 40   // LDS K-stride in halfs: 32 data + 8 pad (80B rows keep 16B alignment)

union F16Frag { v16h v; v8h h[2]; };

// A fragment: 16x32 f16. lanes 0-15: row M=lane, K 0..7 & 16..23; lanes 16-31: K 8..15 & 24..31
__device__ __forceinline__ v16h frag_a(const _Float16* As, int m0, int lane) {
  const _Float16* p = As + (m0 + (lane & 15)) * KS + ((lane >> 4) << 3);
  F16Frag f;
  f.h[0] = *(const v8h*)(p);
  f.h[1] = *(const v8h*)(p + 16);
  return f.v;
}

// B fragment: 32x16 f16 from transposed LDS tile Bt[n][k]. lane&15 = N; lanes 0-15: K 0..7 & 16..23; lanes 16-31: K 8..15 & 24..31
__device__ __forceinline__ v16h frag_b(const _Float16* Bt, int n0, int lane) {
  const _Float16* p = Bt + (n0 + (lane & 15)) * KS + ((lane >> 4) << 3);
  F16Frag f;
  f.h[0] = *(const v8h*)(p);
  f.h[1] = *(const v8h*)(p + 16);
  return f.v;
}

__device__ __forceinline__ v8f wmma16(v16h a, v16h b, v8f c) {
  return __builtin_amdgcn_wmma_f32_16x16x32_f16(false, a, false, b, (short)0, c, false, false);
}

// ---------------------------------------------------------------------------
// K1/K7 family: C[4096 x 1024] = X[4096 x 1024] @ W[1024 x 1024]; f32 in, f16 out
// Q/K/V projection: out = (x@W + b) * scale, stored as f16 [bh][s][d]
// ---------------------------------------------------------------------------
__global__ __launch_bounds__(256) void k_proj(
    const float* __restrict__ X, const float* __restrict__ W,
    const float* __restrict__ bias, _Float16* __restrict__ Yh, float scale)
{
  __shared__ __align__(16) _Float16 As[128 * KS];
  __shared__ __align__(16) _Float16 Bt[128 * KS];
  const int tid  = threadIdx.x;
  const int lane = tid & 31;
  const int wave = tid >> 5;
  const int wm   = wave >> 1;          // 0..3 : 32-row strip
  const int wn   = wave & 1;           // 0..1 : 64-col strip
  const int m0b  = blockIdx.x * 128;
  const int n0b  = blockIdx.y * 128;

  v8f acc[2][4] = {};
  const int am = tid >> 1, aseg = (tid & 1) << 4;   // A: row, 16-float segment
  const int bk = tid >> 3, bn   = (tid & 7) << 4;   // B: k-row, 16-col segment

  for (int k0 = 0; k0 < 1024; k0 += 32) {
    { // stage A (128x32), fp32 -> f16
      const float* src = X + (size_t)(m0b + am) * 1024 + k0 + aseg;
      __builtin_prefetch(src + 32, 0, 1);
      v4f f0 = ((const v4f*)src)[0], f1 = ((const v4f*)src)[1];
      v4f f2 = ((const v4f*)src)[2], f3 = ((const v4f*)src)[3];
      _Float16* d = &As[am * KS + aseg];
#pragma unroll
      for (int i = 0; i < 4; ++i) {
        d[i]      = (_Float16)f0[i]; d[4 + i]  = (_Float16)f1[i];
        d[8 + i]  = (_Float16)f2[i]; d[12 + i] = (_Float16)f3[i];
      }
    }
    { // stage Bt[n][k] = W[k0+k][n0b+n], fp32 -> f16, transposed
      const float* src = W + (size_t)(k0 + bk) * 1024 + n0b + bn;
      __builtin_prefetch(src + 32 * 1024, 0, 1);
      v4f f0 = ((const v4f*)src)[0], f1 = ((const v4f*)src)[1];
      v4f f2 = ((const v4f*)src)[2], f3 = ((const v4f*)src)[3];
#pragma unroll
      for (int i = 0; i < 4; ++i) {
        Bt[(bn + i)      * KS + bk] = (_Float16)f0[i];
        Bt[(bn + 4 + i)  * KS + bk] = (_Float16)f1[i];
        Bt[(bn + 8 + i)  * KS + bk] = (_Float16)f2[i];
        Bt[(bn + 12 + i) * KS + bk] = (_Float16)f3[i];
      }
    }
    __syncthreads();
#pragma unroll
    for (int mi = 0; mi < 2; ++mi) {
      v16h a = frag_a(As, wm * 32 + mi * 16, lane);
#pragma unroll
      for (int ni = 0; ni < 4; ++ni) {
        v16h b = frag_b(Bt, wn * 64 + ni * 16, lane);
        acc[mi][ni] = wmma16(a, b, acc[mi][ni]);
      }
    }
    __syncthreads();
  }

  const int hi8 = (lane >> 4) << 3, nc = lane & 15;
#pragma unroll
  for (int mi = 0; mi < 2; ++mi)
#pragma unroll
    for (int ni = 0; ni < 4; ++ni)
#pragma unroll
      for (int r = 0; r < 8; ++r) {
        int row = m0b + wm * 32 + mi * 16 + r + hi8;
        int col = n0b + wn * 64 + ni * 16 + nc;
        float v = (acc[mi][ni][r] + bias[col]) * scale;
        int b = row >> 10, s = row & 1023, h = col >> 6, d = col & 63;
        Yh[((size_t)(((b << 4) | h) << 10 | s) << 6) + d] = (_Float16)v;
      }
}

// ---------------------------------------------------------------------------
// K2: content logits per (b,h): logits[s,t] = q[s,:]·k[t,:] + attn_bias
// ---------------------------------------------------------------------------
__global__ __launch_bounds__(256) void k_logits(
    const _Float16* __restrict__ Qh, const _Float16* __restrict__ Kh,
    const float* __restrict__ attn_bias, float* __restrict__ logits)
{
  __shared__ __align__(16) _Float16 As[128 * KS];
  __shared__ __align__(16) _Float16 Bt[128 * KS];
  const int tid = threadIdx.x, lane = tid & 31, wave = tid >> 5;
  const int wm = wave >> 1, wn = wave & 1;
  const int m0b = blockIdx.x * 128, n0b = blockIdx.y * 128;
  const int bh = blockIdx.z;
  const _Float16* Q = Qh + ((size_t)bh << 16);
  const _Float16* K = Kh + ((size_t)bh << 16);

  v8f acc[2][4] = {};
  const int cm = tid >> 1, cseg = (tid & 1) << 4;

#pragma unroll
  for (int k0 = 0; k0 < 64; k0 += 32) {
    const _Float16* qs = Q + (size_t)(m0b + cm) * 64 + k0 + cseg;
    const _Float16* ks = K + (size_t)(n0b + cm) * 64 + k0 + cseg;
    *(v8h*)&As[cm * KS + cseg]     = ((const v8h*)qs)[0];
    *(v8h*)&As[cm * KS + cseg + 8] = ((const v8h*)qs)[1];
    *(v8h*)&Bt[cm * KS + cseg]     = ((const v8h*)ks)[0];  // Bt[n][k] = k[t][d]
    *(v8h*)&Bt[cm * KS + cseg + 8] = ((const v8h*)ks)[1];
    __syncthreads();
#pragma unroll
    for (int mi = 0; mi < 2; ++mi) {
      v16h a = frag_a(As, wm * 32 + mi * 16, lane);
#pragma unroll
      for (int ni = 0; ni < 4; ++ni) {
        v16h b = frag_b(Bt, wn * 64 + ni * 16, lane);
        acc[mi][ni] = wmma16(a, b, acc[mi][ni]);
      }
    }
    __syncthreads();
  }

  const int hi8 = (lane >> 4) << 3, nc = lane & 15;
#pragma unroll
  for (int mi = 0; mi < 2; ++mi)
#pragma unroll
    for (int ni = 0; ni < 4; ++ni)
#pragma unroll
      for (int r = 0; r < 8; ++r) {
        int row = m0b + wm * 32 + mi * 16 + r + hi8;
        int col = n0b + wn * 64 + ni * 16 + nc;
        size_t idx = ((size_t)bh << 20) + ((size_t)row << 10) + col;
        logits[idx] = acc[mi][ni][r] + attn_bias[idx];
      }
}

// ---------------------------------------------------------------------------
// K3: edge-key logits, batched over bh for fixed s:
//     logits[bh, s, t] += Q_s[bh,:] · edges_key[s,t,:]
// ---------------------------------------------------------------------------
__global__ __launch_bounds__(256) void k_edgek(
    const _Float16* __restrict__ Qh, const float* __restrict__ EK,
    float* __restrict__ logits)
{
  __shared__ __align__(16) _Float16 As[64 * KS];
  __shared__ __align__(16) _Float16 Bt[128 * KS];
  const int tid = threadIdx.x, lane = tid & 31, wave = tid >> 5;
  const int wm = wave >> 1, wn = wave & 1;   // 4 x 2 waves, each 16x64
  const int t0 = blockIdx.x * 128;
  const int s  = blockIdx.y;

  v8f acc[4] = {};
  const int am = tid >> 2, aseg = (tid & 3) << 3;
  const int bn = tid >> 1, bseg = (tid & 1) << 4;

#pragma unroll
  for (int k0 = 0; k0 < 64; k0 += 32) {
    // A[bh][d] = Q[bh][s][d]  (gathered across heads/batch)
    *(v8h*)&As[am * KS + aseg] =
        *(const v8h*)(Qh + ((((size_t)am << 10) + s) << 6) + k0 + aseg);
    { // Bt[t][d] = edges_key[s][t0+t][d], fp32 -> f16
      const float* src = EK + ((((size_t)s << 10) + t0 + bn) << 6) + k0 + bseg;
      __builtin_prefetch(src + 32, 0, 1);
      v4f f0 = ((const v4f*)src)[0], f1 = ((const v4f*)src)[1];
      v4f f2 = ((const v4f*)src)[2], f3 = ((const v4f*)src)[3];
      _Float16* d = &Bt[bn * KS + bseg];
#pragma unroll
      for (int i = 0; i < 4; ++i) {
        d[i]      = (_Float16)f0[i]; d[4 + i]  = (_Float16)f1[i];
        d[8 + i]  = (_Float16)f2[i]; d[12 + i] = (_Float16)f3[i];
      }
    }
    __syncthreads();
    v16h a = frag_a(As, wm * 16, lane);
#pragma unroll
    for (int ni = 0; ni < 4; ++ni) {
      v16h b = frag_b(Bt, wn * 64 + ni * 16, lane);
      acc[ni] = wmma16(a, b, acc[ni]);
    }
    __syncthreads();
  }

  const int hi8 = (lane >> 4) << 3, nc = lane & 15;
#pragma unroll
  for (int ni = 0; ni < 4; ++ni)
#pragma unroll
    for (int r = 0; r < 8; ++r) {
      int bhrow = wm * 16 + r + hi8;
      int col = t0 + wn * 64 + ni * 16 + nc;
      float* p = logits + ((size_t)bhrow << 20) + ((size_t)s << 10) + col;
      *p += acc[ni][r];
    }
}

// ---------------------------------------------------------------------------
// K4: softmax over T per (bh,s) row; writes f16 weights IN PLACE over the
//     fp32 row (first 2KB of each 4KB row) -> halves A traffic for W-GEMMs.
// ---------------------------------------------------------------------------
__global__ __launch_bounds__(256) void k_softmax(float* __restrict__ logits)
{
  const int lane = threadIdx.x & 31;
  const size_t row = (size_t)blockIdx.x * 8 + (threadIdx.x >> 5);
  float* p = logits + (row << 10);
  float v[32];
  const v4f* src = (const v4f*)(p + (lane << 5));
  float mx = -3.0e38f;
#pragma unroll
  for (int i = 0; i < 8; ++i) {
    v4f t = src[i];
#pragma unroll
    for (int j = 0; j < 4; ++j) { v[i * 4 + j] = t[j]; mx = fmaxf(mx, t[j]); }
  }
#pragma unroll
  for (int off = 16; off > 0; off >>= 1) mx = fmaxf(mx, __shfl_xor(mx, off, 32));
  float sum = 0.f;
#pragma unroll
  for (int i = 0; i < 32; ++i) { v[i] = __expf(v[i] - mx); sum += v[i]; }
#pragma unroll
  for (int off = 16; off > 0; off >>= 1) sum += __shfl_xor(sum, off, 32);
  const float inv = 1.0f / sum;
  _Float16* ph = (_Float16*)p;   // stores issue only after loads completed (data dep)
#pragma unroll
  for (int i = 0; i < 4; ++i) {
    v8h o;
#pragma unroll
    for (int j = 0; j < 8; ++j) o[j] = (_Float16)(v[i * 8 + j] * inv);
    *(v8h*)(ph + (lane << 5) + (i << 3)) = o;
  }
}

// ---------------------------------------------------------------------------
// K5: content output per (b,h): O[s,d] = sum_t w[s,t] * v[t,d]
//     A = f16 weights embedded in logits rows (half-row stride 2048)
// ---------------------------------------------------------------------------
__global__ __launch_bounds__(256) void k_wv(
    const float* __restrict__ logits, const _Float16* __restrict__ Vh,
    float* __restrict__ O)
{
  __shared__ __align__(16) _Float16 As[128 * KS];
  __shared__ __align__(16) _Float16 Bt[64 * KS];
  const int tid = threadIdx.x, lane = tid & 31, wave = tid >> 5;
  const int m0b = blockIdx.x * 128;
  const int bh  = blockIdx.y;
  const _Float16* Wh = (const _Float16*)logits + ((size_t)bh << 21);
  const _Float16* V  = Vh + ((size_t)bh << 16);

  v8f acc[4] = {};
  const int am = tid >> 1, aseg = (tid & 1) << 4;
  const int bk = tid >> 3, bn = (tid & 7) << 3;

  for (int k0 = 0; k0 < 1024; k0 += 32) {
    const _Float16* asrc = Wh + ((size_t)(m0b + am) << 11) + k0 + aseg;
    *(v8h*)&As[am * KS + aseg]     = ((const v8h*)asrc)[0];
    *(v8h*)&As[am * KS + aseg + 8] = ((const v8h*)asrc)[1];
    { // Bt[d][t] = v[t][d]  (transpose stage)
      v8h t = *(const v8h*)(V + ((size_t)(k0 + bk) << 6) + bn);
#pragma unroll
      for (int i = 0; i < 8; ++i) Bt[(bn + i) * KS + bk] = t[i];
    }
    __syncthreads();
    v16h a = frag_a(As, wave * 16, lane);
#pragma unroll
    for (int ni = 0; ni < 4; ++ni) {
      v16h b = frag_b(Bt, ni * 16, lane);
      acc[ni] = wmma16(a, b, acc[ni]);
    }
    __syncthreads();
  }

  const int hi8 = (lane >> 4) << 3, nc = lane & 15;
#pragma unroll
  for (int ni = 0; ni < 4; ++ni)
#pragma unroll
    for (int r = 0; r < 8; ++r) {
      int row = m0b + wave * 16 + r + hi8;
      int col = ni * 16 + nc;
      O[((((size_t)bh << 10) + row) << 6) + col] = acc[ni][r];
    }
}

// ---------------------------------------------------------------------------
// K6: edge-value, batched over bh for fixed s:
//     O[bh, s, d] += sum_t w[bh,s,t] * edges_value[s,t,d]
// ---------------------------------------------------------------------------
__global__ __launch_bounds__(256) void k_edgev(
    const float* __restrict__ logits, const float* __restrict__ EV,
    float* __restrict__ O)
{
  __shared__ __align__(16) _Float16 As[64 * KS];
  __shared__ __align__(16) _Float16 Bt[64 * KS];
  const int tid = threadIdx.x, lane = tid & 31, wave = tid >> 5;
  const int wm = wave >> 1, wn = wave & 1;   // 4 x 2 waves, each 16x32
  const int s = blockIdx.x;
  const _Float16* Wh = (const _Float16*)logits;

  v8f acc[2] = {};
  const int am = tid >> 2, aseg = (tid & 3) << 3;
  const int bk = tid >> 3, bn = (tid & 7) << 3;

  for (int k0 = 0; k0 < 1024; k0 += 32) {
    // A[bh][t] = w_f16[bh][s][t]
    *(v8h*)&As[am * KS + aseg] =
        *(const v8h*)(Wh + ((((size_t)am << 10) + s) << 11) + k0 + aseg);
    { // Bt[d][t] = edges_value[s][t][d], fp32 -> f16, transposed
      const float* src = EV + ((((size_t)s << 10) + k0 + bk) << 6) + bn;
      __builtin_prefetch(src + 32 * 64, 0, 1);
      v4f f0 = ((const v4f*)src)[0], f1 = ((const v4f*)src)[1];
#pragma unroll
      for (int i = 0; i < 4; ++i) {
        Bt[(bn + i)     * KS + bk] = (_Float16)f0[i];
        Bt[(bn + 4 + i) * KS + bk] = (_Float16)f1[i];
      }
    }
    __syncthreads();
    v16h a = frag_a(As, wm * 16, lane);
#pragma unroll
    for (int ni = 0; ni < 2; ++ni) {
      v16h b = frag_b(Bt, wn * 32 + ni * 16, lane);
      acc[ni] = wmma16(a, b, acc[ni]);
    }
    __syncthreads();
  }

  const int hi8 = (lane >> 4) << 3, nc = lane & 15;
#pragma unroll
  for (int ni = 0; ni < 2; ++ni)
#pragma unroll
    for (int r = 0; r < 8; ++r) {
      int bhrow = wm * 16 + r + hi8;
      int col = wn * 32 + ni * 16 + nc;
      float* p = O + ((((size_t)bhrow << 10) + s) << 6) + col;
      *p += acc[ni][r];
    }
}

// ---------------------------------------------------------------------------
// K7: output projection Y[4096,1024] = headcombine(O) @ Wp + bp
// ---------------------------------------------------------------------------
__global__ __launch_bounds__(256) void k_out(
    const float* __restrict__ O, const float* __restrict__ Wp,
    const float* __restrict__ bp, float* __restrict__ Y)
{
  __shared__ __align__(16) _Float16 As[128 * KS];
  __shared__ __align__(16) _Float16 Bt[128 * KS];
  const int tid = threadIdx.x, lane = tid & 31, wave = tid >> 5;
  const int wm = wave >> 1, wn = wave & 1;
  const int m0b = blockIdx.x * 128, n0b = blockIdx.y * 128;

  v8f acc[2][4] = {};
  const int am = tid >> 1, aseg = (tid & 1) << 4;
  const int bk = tid >> 3, bn = (tid & 7) << 4;

  for (int k0 = 0; k0 < 1024; k0 += 32) {
    { // A[m][e] = O[b, h, s, d] with m=(b,s), e=(h,d); 16-chunk stays inside one head
      int m = m0b + am, e = k0 + aseg;
      int b = m >> 10, s = m & 1023, h = e >> 6, d0 = e & 63;
      const float* src = O + ((((size_t)((b << 4) | h) << 10) + s) << 6) + d0;
      v4f f0 = ((const v4f*)src)[0], f1 = ((const v4f*)src)[1];
      v4f f2 = ((const v4f*)src)[2], f3 = ((const v4f*)src)[3];
      _Float16* d = &As[am * KS + aseg];
#pragma unroll
      for (int i = 0; i < 4; ++i) {
        d[i]      = (_Float16)f0[i]; d[4 + i]  = (_Float16)f1[i];
        d[8 + i]  = (_Float16)f2[i]; d[12 + i] = (_Float16)f3[i];
      }
    }
    { // Bt[n][k] = Wp[k][n]
      const float* src = Wp + (size_t)(k0 + bk) * 1024 + n0b + bn;
      __builtin_prefetch(src + 32 * 1024, 0, 1);
      v4f f0 = ((const v4f*)src)[0], f1 = ((const v4f*)src)[1];
      v4f f2 = ((const v4f*)src)[2], f3 = ((const v4f*)src)[3];
#pragma unroll
      for (int i = 0; i < 4; ++i) {
        Bt[(bn + i)      * KS + bk] = (_Float16)f0[i];
        Bt[(bn + 4 + i)  * KS + bk] = (_Float16)f1[i];
        Bt[(bn + 8 + i)  * KS + bk] = (_Float16)f2[i];
        Bt[(bn + 12 + i) * KS + bk] = (_Float16)f3[i];
      }
    }
    __syncthreads();
#pragma unroll
    for (int mi = 0; mi < 2; ++mi) {
      v16h a = frag_a(As, wm * 32 + mi * 16, lane);
#pragma unroll
      for (int ni = 0; ni < 4; ++ni) {
        v16h b = frag_b(Bt, wn * 64 + ni * 16, lane);
        acc[mi][ni] = wmma16(a, b, acc[mi][ni]);
      }
    }
    __syncthreads();
  }

  const int hi8 = (lane >> 4) << 3, nc = lane & 15;
#pragma unroll
  for (int mi = 0; mi < 2; ++mi)
#pragma unroll
    for (int ni = 0; ni < 4; ++ni)
#pragma unroll
      for (int r = 0; r < 8; ++r) {
        int row = m0b + wm * 32 + mi * 16 + r + hi8;
        int col = n0b + wn * 64 + ni * 16 + nc;
        Y[((size_t)row << 10) + col] = acc[mi][ni][r] + bp[col];
      }
}

// ---------------------------------------------------------------------------
extern "C" void kernel_launch(void* const* d_in, const int* in_sizes, int n_in,
                              void* d_out, int out_size, void* d_ws, size_t ws_size,
                              hipStream_t stream)
{
  (void)in_sizes; (void)n_in; (void)out_size; (void)ws_size;
  const float* queries = (const float*)d_in[0];
  const float* keys    = (const float*)d_in[1];
  const float* values  = (const float*)d_in[2];
  const float* ek      = (const float*)d_in[3];
  const float* ev      = (const float*)d_in[4];
  const float* abias   = (const float*)d_in[5];
  const float* Wq = (const float*)d_in[6];  const float* bq = (const float*)d_in[7];
  const float* Wk = (const float*)d_in[8];  const float* bk = (const float*)d_in[9];
  const float* Wv = (const float*)d_in[10]; const float* bv = (const float*)d_in[11];
  const float* Wp = (const float*)d_in[12]; const float* bp = (const float*)d_in[13];

  // workspace layout
  const size_t LOGITS_B = (size_t)64 * 1024 * 1024 * 4;  // 256 MB fp32 [bh][s][t]
  const size_t H16_B    = (size_t)64 * 1024 * 64 * 2;    // 8 MB f16 [bh][s][d]
  char* ws = (char*)d_ws;
  float*    logits = (float*)ws;
  _Float16* qh = (_Float16*)(ws + LOGITS_B);
  _Float16* kh = (_Float16*)(ws + LOGITS_B + H16_B);
  _Float16* vh = (_Float16*)(ws + LOGITS_B + 2 * H16_B);
  float*    O  = (float*)(ws + LOGITS_B + 3 * H16_B);    // 16 MB fp32 [bh][s][d]

  // Q/K/V projections (Q pre-scaled by D^-0.5 = 0.125)
  k_proj<<<dim3(32, 8), 256, 0, stream>>>(queries, Wq, bq, qh, 0.125f);
  k_proj<<<dim3(32, 8), 256, 0, stream>>>(keys,    Wk, bk, kh, 1.0f);
  k_proj<<<dim3(32, 8), 256, 0, stream>>>(values,  Wv, bv, vh, 1.0f);
  // content logits + attn_bias
  k_logits<<<dim3(8, 8, 64), 256, 0, stream>>>(qh, kh, abias, logits);
  // edge-key logits (per-s GEMM batched over bh)
  k_edgek<<<dim3(8, 1024), 256, 0, stream>>>(qh, ek, logits);
  // softmax -> f16 weights in place
  k_softmax<<<dim3(8192), 256, 0, stream>>>(logits);
  // w @ v
  k_wv<<<dim3(8, 64), 256, 0, stream>>>(logits, vh, O);
  // edge-value accumulation (per-s GEMM batched over bh)
  k_edgev<<<dim3(1024), 256, 0, stream>>>(logits, ev, O);
  // output projection
  k_out<<<dim3(32, 8), 256, 0, stream>>>(O, Wp, bp, (float*)d_out);
}
